// RecurrentMemoryTransformer_10969346474830
// MI455X (gfx1250) — matrix operand order repaired and checked by reading, compiled, not run on hardware
//
#include <hip/hip_runtime.h>
#include <hip/hip_bf16.h>

// ---------------- model constants ----------------
#define DIMM   1024
#define HEADS  16
#define DH     64
#define DEPTH  6
#define VOCAB  32000
#define SEQN   1024
#define MEMT   64
#define LTOT   (SEQN + MEMT)          // 1088
#define DII    2730
#define DII_PAD 2736                  // 16B-aligned row stride for ag buffer
#define THETA  32768.0f

typedef __bf16 bf16_t;
typedef __attribute__((ext_vector_type(16))) __bf16 v16bf;
typedef __attribute__((ext_vector_type(8)))  __bf16 v8bf;
typedef __attribute__((ext_vector_type(8)))  float  v8f;

union PK32 { unsigned int u; __bf16 h[2]; };

#if defined(__HIP_DEVICE_COMPILE__) && __has_builtin(__builtin_amdgcn_global_load_async_to_lds_b128)
#define HAVE_ASYNC_LDS 1
#else
#define HAVE_ASYNC_LDS 0
#endif

#if HAVE_ASYNC_LDS
typedef int v4i_t __attribute__((vector_size(16)));
typedef v4i_t __attribute__((address_space(1))) gv4i_t;   // global  (prints __device__)
typedef v4i_t __attribute__((address_space(3))) sv4i_t;   // LDS     (prints __shared__)
#endif

// async 16-byte global -> LDS copy (ASYNCcnt); falls back to register copy
__device__ __forceinline__ void async_cp16(const bf16_t* g, bf16_t* l) {
#if HAVE_ASYNC_LDS
    __builtin_amdgcn_global_load_async_to_lds_b128(
        (gv4i_t*)(void*)g, (sv4i_t*)(void*)l, 0, 0);
#else
    *(v8bf*)l = *(const v8bf*)g;
#endif
}
__device__ __forceinline__ void wait_async() {
#if HAVE_ASYNC_LDS
#if __has_builtin(__builtin_amdgcn_s_wait_asynccnt)
    __builtin_amdgcn_s_wait_asynccnt(0);
#else
    asm volatile("s_wait_asynccnt 0x0" ::: "memory");
#endif
#endif
}

// =====================================================================
// WMMA GEMM:  C[M x N] = A(bf16)[M x K] * B + (optional) Cin
// BMODE 0: B is fp32 row-major [K x ldb]   (weights / V)
// BMODE 1: B is bf16 "transposed": element (k,n) = B[n*ldb + k]  (K^T)
// CMODE 0: C fp32 (ldc), optional fp32 residual Cin (ldcin)
// CMODE 1: C bf16 (ldc), no residual
// Block: 256 threads (8 waves). Tile: BM=128, BN=64, BK=32.
// LDS tiles stored in fragment order; row stride 40 bf16 (80B = 20 banks)
// => conflict-free, 16B-aligned ds_load_b128 fragment reads.
// A tile (and B tile in BMODE1) staged via GLOBAL_LOAD_ASYNC_TO_LDS_B128,
// per-lane LDS destination addresses keep natural K order:
//  K-block b (8 bf16 = 16B) -> position block b  (0,1,2,3).
// =====================================================================
template <int BMODE, int CMODE>
__global__ __launch_bounds__(256)
void gemm_wmma_bf16(const bf16_t* __restrict__ A, int lda,
                    const void* __restrict__ Bv, int ldb,
                    const float* __restrict__ Cin, int ldcin,
                    void* __restrict__ Cout, int ldc,
                    int M, int N, int K)
{
    constexpr int BM = 128, BN = 64, BK = 32;
    constexpr int AS = 40;                  // row stride (bf16) for both tiles
    __shared__ bf16_t As[BM * AS];          // [row][K 0..31]      10.0 KB
    __shared__ bf16_t Bs[BN * AS];          // [col n][K 0..31]        5.0 KB

    const int tid  = threadIdx.x;
    const int wave = tid >> 5;
    const int lane = tid & 31;
    const int half = lane >> 4;
    const int l16  = lane & 15;
    const int bm0  = blockIdx.y * BM;
    const int bn0  = blockIdx.x * BN;

    const float*  Bf = (const float*)Bv;
    const bf16_t* Bh = (const bf16_t*)Bv;

    v8f acc[4];
    v8f zero = {0.f,0.f,0.f,0.f,0.f,0.f,0.f,0.f};
    acc[0] = zero; acc[1] = zero; acc[2] = zero; acc[3] = zero;

    for (int k0 = 0; k0 < K; k0 += BK) {
        // ---- stage A tile: 128 rows x 4 chunks of 16B, async scatter ----
        for (int i = tid; i < BM * 4; i += 256) {
            int r   = i >> 2;
            int blk = i & 3;
            int pbk = blk;                               // 0,1,2,3
            int gr  = bm0 + r, gk = k0 + blk * 8;
            bf16_t* dst = &As[r * AS + pbk * 8];
            if (gr < M && gk + 8 <= K) {
                async_cp16(A + (size_t)gr * lda + gk, dst);
            } else {
#pragma unroll
                for (int e = 0; e < 8; ++e)
                    dst[e] = (gr < M && gk + e < K) ? A[(size_t)gr * lda + gk + e]
                                                    : (bf16_t)0.f;
            }
        }
        // ---- stage B tile: 32x64 -> Bs[n][k] (column-major, natural K) ----
        if (BMODE == 0) {
            // fp32 row-major: coalesced reads along N, cvt, b16 LDS stores
            for (int i = tid; i < BK * BN / 2; i += 256) {
                int kr = i >> 5;
                int cp = (i & 31) * 2;
                int gk = k0 + kr, gn = bn0 + cp;
                float b0 = 0.f, b1 = 0.f;
                if (gk < K) {
                    if (gn     < N) b0 = Bf[(size_t)gk * ldb + gn];
                    if (gn + 1 < N) b1 = Bf[(size_t)gk * ldb + gn + 1];
                    if (gk + BK < K && gn < N)   // prefetch next K-step rows
                        __builtin_prefetch(Bf + (size_t)(gk + BK) * ldb + gn, 0, 0);
                }
                Bs[(cp)     * AS + kr] = (bf16_t)b0;
                Bs[(cp + 1) * AS + kr] = (bf16_t)b1;
            }
        } else {
            // bf16 transposed: contiguous along K -> async 16B chunks
            for (int i = tid; i < BN * 4; i += 256) {
                int n   = i >> 2;
                int blk = i & 3;
                int gn  = bn0 + n, gk = k0 + blk * 8;
                bf16_t* dst = &Bs[n * AS + blk * 8];
                if (gn < N && gk + 8 <= K) {
                    async_cp16(Bh + (size_t)gn * ldb + gk, dst);
                } else {
#pragma unroll
                    for (int e = 0; e < 8; ++e)
                        dst[e] = (gn < N && gk + e < K) ? Bh[(size_t)gn * ldb + gk + e]
                                                        : (bf16_t)0.f;
                }
            }
        }
        wait_async();
        __syncthreads();

        // ---- A fragment: two ds_load_b128, contiguous & conflict-free ----
        v16bf afrag;
        {
            const v8bf* pa = (const v8bf*)&As[((wave << 4) + l16) * AS + half * 16];
            v8bf a0 = pa[0], a1 = pa[1];
            afrag = __builtin_shufflevector(a0, a1, 0,1,2,3,4,5,6,7,8,9,10,11,12,13,14,15);
        }
        // ---- load all 4 B fragments, then 4 back-to-back WMMAs ----
        v16bf bfrag[4];
#pragma unroll
        for (int c = 0; c < 4; ++c) {
            const v8bf* pb = (const v8bf*)&Bs[((c << 4) + l16) * AS + half * 16];
            v8bf b0 = pb[0], b1 = pb[1];
            bfrag[c] = __builtin_shufflevector(b0, b1, 0,1,2,3,4,5,6,7,8,9,10,11,12,13,14,15);
        }
#pragma unroll
        for (int c = 0; c < 4; ++c) {
            acc[c] = __builtin_amdgcn_wmma_f32_16x16x32_bf16(
                false, afrag, false, bfrag[c], (short)0, acc[c], false, false);
        }
        __syncthreads();
    }

    // ---- epilogue: C/D layout row = v + 8*half, col = lane&15 ----
#pragma unroll
    for (int c = 0; c < 4; ++c) {
#pragma unroll
        for (int v = 0; v < 8; ++v) {
            int gr = bm0 + (wave << 4) + v + 8 * half;
            int gn = bn0 + (c << 4) + l16;
            if (gr < M && gn < N) {
                float val = acc[c][v];
                if (CMODE == 0) {
                    if (Cin) val += Cin[(size_t)gr * ldcin + gn];
                    ((float*)Cout)[(size_t)gr * ldc + gn] = val;
                } else {
                    ((bf16_t*)Cout)[(size_t)gr * ldc + gn] = (bf16_t)val;
                }
            }
        }
    }
}

// =====================================================================
// elementwise kernels
// =====================================================================
__global__ void embed_kernel(const int* __restrict__ x,
                             const float* __restrict__ tok,
                             const float* __restrict__ pos,
                             const float* __restrict__ mem,
                             float* __restrict__ h)
{
    size_t idx = (size_t)blockIdx.x * blockDim.x + threadIdx.x;
    if (idx >= (size_t)LTOT * DIMM) return;
    int r = (int)(idx / DIMM), d = (int)(idx % DIMM);
    float v;
    if (r < SEQN) v = tok[(size_t)x[r] * DIMM + d] + pos[(size_t)r * DIMM + d];
    else          v = mem[(size_t)(r - SEQN) * DIMM + d];
    h[idx] = v;
}

__global__ __launch_bounds__(256)
void rmsnorm_bf16_kernel(const float* __restrict__ X,
                         const float* __restrict__ g,
                         bf16_t* __restrict__ Y, int D)
{
    int row = blockIdx.x;
    const float* x = X + (size_t)row * D;
    __shared__ float red[256];
    float s = 0.f;
    for (int j = threadIdx.x; j < D; j += 256) { float v = x[j]; s += v * v; }
    red[threadIdx.x] = s; __syncthreads();
    for (int t = 128; t > 0; t >>= 1) {
        if (threadIdx.x < t) red[threadIdx.x] += red[threadIdx.x + t];
        __syncthreads();
    }
    float f = sqrtf((float)D) / fmaxf(sqrtf(red[0]), 1e-12f);
    for (int j = threadIdx.x; j < D; j += 256)
        Y[(size_t)row * D + j] = (bf16_t)(x[j] * f * g[j]);
}

// RoPE: read fp32 head-major [row, head*64 + d] with leading dim lds,
// write rotated bf16 with leading dim ldd. One thread per (row, head, i<32).
__global__ void rope_kernel(const float* __restrict__ src, int lds,
                            bf16_t* __restrict__ dst, int ldd)
{
    int idx = blockIdx.x * blockDim.x + threadIdx.x;   // LTOT*HEADS*32
    if (idx >= LTOT * HEADS * 32) return;
    int i   = idx & 31;
    int hd  = (idx >> 5) & (HEADS - 1);
    int row = idx >> 9;
    float p   = (row < SEQN) ? ((float)row + 10000.0f) : 0.0f;
    float inv = __powf(THETA, -(2.0f * (float)i) / (float)DH);
    float a   = p * inv;
    float c = __cosf(a), s = __sinf(a);
    const float* q = src + (size_t)row * lds + hd * DH;
    float x0 = q[i], x1 = q[i + 32];
    bf16_t* o = dst + (size_t)row * ldd + hd * DH;
    o[i]      = (bf16_t)(x0 * c - x1 * s);
    o[i + 32] = (bf16_t)(x1 * c + x0 * s);
}

// masked softmax over one row of scores (L cols), scale DH^-0.5, write bf16
__global__ __launch_bounds__(256)
void softmax_mask_kernel(const float* __restrict__ S, bf16_t* __restrict__ P)
{
    int row = blockIdx.x;
    const float scale = 0.125f;   // 64^-0.5
    __shared__ float red[256];
    float mx = -3.4e38f;
    for (int j = threadIdx.x; j < LTOT; j += 256) {
        bool ok = (row >= SEQN) || (j >= SEQN) || (j <= row);
        if (ok) mx = fmaxf(mx, S[(size_t)row * LTOT + j] * scale);
    }
    red[threadIdx.x] = mx; __syncthreads();
    for (int t = 128; t > 0; t >>= 1) {
        if (threadIdx.x < t) red[threadIdx.x] = fmaxf(red[threadIdx.x], red[threadIdx.x + t]);
        __syncthreads();
    }
    mx = red[0]; __syncthreads();
    float sum = 0.f;
    for (int j = threadIdx.x; j < LTOT; j += 256) {
        bool ok = (row >= SEQN) || (j >= SEQN) || (j <= row);
        if (ok) sum += __expf(S[(size_t)row * LTOT + j] * scale - mx);
    }
    red[threadIdx.x] = sum; __syncthreads();
    for (int t = 128; t > 0; t >>= 1) {
        if (threadIdx.x < t) red[threadIdx.x] += red[threadIdx.x + t];
        __syncthreads();
    }
    float inv = 1.0f / red[0];
    for (int j = threadIdx.x; j < LTOT; j += 256) {
        bool ok = (row >= SEQN) || (j >= SEQN) || (j <= row);
        float e = ok ? __expf(S[(size_t)row * LTOT + j] * scale - mx) * inv : 0.f;
        P[(size_t)row * LTOT + j] = (bf16_t)e;
    }
}

// GEGLU: u[L x 5460] -> ag[L x DII_PAD] bf16, ag = a * gelu_exact(gate),
// pad columns [2730, 2736) zero-filled so the GEMM A tile stays clean.
__global__ void geglu_kernel(const float* __restrict__ U, bf16_t* __restrict__ AG)
{
    size_t idx = (size_t)blockIdx.x * blockDim.x + threadIdx.x;
    if (idx >= (size_t)LTOT * DII_PAD) return;
    int r = (int)(idx / DII_PAD), c = (int)(idx % DII_PAD);
    float out = 0.f;
    if (c < DII) {
        float a = U[(size_t)r * (2 * DII) + c];
        float g = U[(size_t)r * (2 * DII) + DII + c];
        out = a * (0.5f * g * (1.0f + erff(g * 0.70710678118f)));
    }
    AG[idx] = (bf16_t)out;
}

__global__ void copy_f32_kernel(const float* __restrict__ src,
                                float* __restrict__ dst, size_t count)
{
    size_t idx = (size_t)blockIdx.x * blockDim.x + threadIdx.x;
    if (idx < count) dst[idx] = src[idx];
}

// =====================================================================
// host orchestration
// =====================================================================
extern "C" void kernel_launch(void* const* d_in, const int* in_sizes, int n_in,
                              void* d_out, int out_size, void* d_ws, size_t ws_size,
                              hipStream_t stream)
{
    const int*   x          = (const int*)  d_in[0];
    const float* token_emb  = (const float*)d_in[1];
    const float* pos_emb    = (const float*)d_in[2];
    const float* mem_tok    = (const float*)d_in[3];
    const float* attn_gamma = (const float*)d_in[4];
    const float* wq         = (const float*)d_in[5];
    const float* wkv        = (const float*)d_in[6];
    const float* wo         = (const float*)d_in[7];
    const float* ff_gamma   = (const float*)d_in[8];
    const float* ff_w1      = (const float*)d_in[9];
    const float* ff_w2      = (const float*)d_in[10];
    const float* out_gamma  = (const float*)d_in[11];
    const float* w_logits   = (const float*)d_in[12];

    // ---- carve workspace ----
    char* ws = (char*)d_ws;
    auto carve = [&](size_t bytes) -> void* {
        void* p = (void*)ws;
        ws += (bytes + 255) & ~(size_t)255;
        return p;
    };
    float*  h      = (float*) carve((size_t)LTOT * DIMM * 4);          // residual stream
    float*  x1     = (float*) carve((size_t)LTOT * DIMM * 4);          // post-attn residual
    bf16_t* xn     = (bf16_t*)carve((size_t)LTOT * DIMM * 2);          // rmsnorm out
    float*  qf     = (float*) carve((size_t)LTOT * DIMM * 4);          // Q fp32
    float*  kvf    = (float*) carve((size_t)LTOT * 2 * DIMM * 4);      // K|V fp32
    bf16_t* qb     = (bf16_t*)carve((size_t)LTOT * DIMM * 2);          // RoPE(Q) bf16
    bf16_t* kb     = (bf16_t*)carve((size_t)LTOT * DIMM * 2);          // RoPE(K) bf16
    float*  sc     = (float*) carve((size_t)LTOT * LTOT * 4);          // scores (per head)
    bf16_t* pb     = (bf16_t*)carve((size_t)LTOT * LTOT * 2);          // probs bf16
    bf16_t* ob     = (bf16_t*)carve((size_t)LTOT * DIMM * 2);          // attn out bf16
    float*  u      = (float*) carve((size_t)LTOT * 2 * DII * 4);       // ffn up
    bf16_t* ag     = (bf16_t*)carve((size_t)LTOT * DII_PAD * 2);       // a*gelu(gate)
    bf16_t* tn     = (bf16_t*)carve((size_t)SEQN * DIMM * 2);          // final norm bf16

    const int T = 256;
    dim3 blk(T);

    // ---- embedding ----
    embed_kernel<<<(LTOT * DIMM + T - 1) / T, blk, 0, stream>>>(x, token_emb, pos_emb, mem_tok, h);

    const int MT = (LTOT + 127) / 128;     // 9 row-tiles for L=1088

    for (int layer = 0; layer < DEPTH; ++layer) {
        const float* wq_l  = wq    + (size_t)layer * DIMM * (HEADS * DH);
        const float* wkv_l = wkv   + (size_t)layer * DIMM * (2 * HEADS * DH);
        const float* wo_l  = wo    + (size_t)layer * (HEADS * DH) * DIMM;
        const float* w1_l  = ff_w1 + (size_t)layer * DIMM * (2 * DII);
        const float* w2_l  = ff_w2 + (size_t)layer * DII * DIMM;

        // xn = rmsnorm(h) -> bf16
        rmsnorm_bf16_kernel<<<LTOT, blk, 0, stream>>>(h, attn_gamma + (size_t)layer * DIMM, xn, DIMM);

        // qf = xn @ wq   (L x 1024 x 1024)
        gemm_wmma_bf16<0, 0><<<dim3(DIMM / 64, MT), blk, 0, stream>>>(
            xn, DIMM, wq_l, DIMM, nullptr, 0, qf, DIMM, LTOT, DIMM, DIMM);
        // kvf = xn @ wkv (L x 2048 x 1024)
        gemm_wmma_bf16<0, 0><<<dim3(2 * DIMM / 64, MT), blk, 0, stream>>>(
            xn, DIMM, wkv_l, 2 * DIMM, nullptr, 0, kvf, 2 * DIMM, LTOT, 2 * DIMM, DIMM);

        // RoPE -> bf16
        rope_kernel<<<(LTOT * HEADS * 32 + T - 1) / T, blk, 0, stream>>>(qf, DIMM, qb, DIMM);
        rope_kernel<<<(LTOT * HEADS * 32 + T - 1) / T, blk, 0, stream>>>(kvf, 2 * DIMM, kb, DIMM);

        // attention per head
        for (int hd = 0; hd < HEADS; ++hd) {
            // scores = q_h @ k_h^T   (L x L x 64), B = bf16 transposed
            gemm_wmma_bf16<1, 0><<<dim3(LTOT / 64, MT), blk, 0, stream>>>(
                qb + hd * DH, DIMM, kb + hd * DH, DIMM, nullptr, 0, sc, LTOT, LTOT, LTOT, DH);
            // masked softmax -> bf16 probs
            softmax_mask_kernel<<<LTOT, blk, 0, stream>>>(sc, pb);
            // o_h = probs @ v_h  (L x 64 x L), B = fp32 slice of kvf, out bf16 strided
            gemm_wmma_bf16<0, 1><<<dim3(1, MT), blk, 0, stream>>>(
                pb, LTOT, kvf + DIMM + hd * DH, 2 * DIMM, nullptr, 0, ob + hd * DH, DIMM,
                LTOT, DH, LTOT);
        }

        // x1 = h + ob @ wo
        gemm_wmma_bf16<0, 0><<<dim3(DIMM / 64, MT), blk, 0, stream>>>(
            ob, DIMM, wo_l, DIMM, h, DIMM, x1, DIMM, LTOT, DIMM, DIMM);

        // FFN
        rmsnorm_bf16_kernel<<<LTOT, blk, 0, stream>>>(x1, ff_gamma + (size_t)layer * DIMM, xn, DIMM);
        gemm_wmma_bf16<0, 0><<<dim3((2 * DII + 63) / 64, MT), blk, 0, stream>>>(
            xn, DIMM, w1_l, 2 * DII, nullptr, 0, u, 2 * DII, LTOT, 2 * DII, DIMM);
        geglu_kernel<<<(int)(((size_t)LTOT * DII_PAD + T - 1) / T), blk, 0, stream>>>(u, ag);
        // h = x1 + ag @ w2  (K = 2730, bounds-checked; lda padded to 2736)
        gemm_wmma_bf16<0, 0><<<dim3(DIMM / 64, MT), blk, 0, stream>>>(
            ag, DII_PAD, w2_l, DIMM, x1, DIMM, h, DIMM, LTOT, DIMM, DII);
    }

    // final norm over token rows only + logits
    rmsnorm_bf16_kernel<<<SEQN, blk, 0, stream>>>(h, out_gamma, tn, DIMM);
    gemm_wmma_bf16<0, 0><<<dim3(VOCAB / 64, SEQN / 128), blk, 0, stream>>>(
        tn, DIMM, w_logits, VOCAB, nullptr, 0, (float*)d_out, VOCAB, SEQN, VOCAB, DIMM);

    // write_out = h[n:, :] appended after logits
    copy_f32_kernel<<<(MEMT * DIMM + T - 1) / T, blk, 0, stream>>>(
        h + (size_t)SEQN * DIMM, (float*)d_out + (size_t)SEQN * VOCAB, (size_t)MEMT * DIMM);
}
